// CTNet_4131758539075
// MI455X (gfx1250) — matrix operand order repaired and checked by reading, not compiled, not run on hardware
//
#include <hip/hip_runtime.h>
#include <math.h>

// ---------------------------------------------------------------------------
// CTNet forward for MI455X (gfx1250). B=16, N=1024, Fin=128, H=32, K1=64,
// K2=16, OUT=10. Heavy N x N work done with v_wmma_f32_16x16x32_f16, fused so
// adj is streamed once per logical pass (HBM-bound design). B-panels for the
// adj_new matmuls are staged into LDS with the Tensor Data Mover
// (tensor_load_to_lds + s_wait_tensorcnt) so the per-K-step fragment gathers
// hit LDS instead of issuing 16 scalar global loads per lane.
// ---------------------------------------------------------------------------

#define Bb   16
#define Nn   1024
#define FIN  128
#define Hh   32
#define K1   64
#define K2   16
#define OUTC 10
#define EPSf 1e-15f

typedef _Float16 half_t;
typedef _Float16 v16h __attribute__((ext_vector_type(16)));
typedef float    v8f  __attribute__((ext_vector_type(8)));
typedef unsigned int v4u  __attribute__((ext_vector_type(4)));
typedef unsigned int u32x4 __attribute__((ext_vector_type(4)));
typedef int          i32x4 __attribute__((ext_vector_type(4)));
typedef int          i32x8 __attribute__((ext_vector_type(8)));

union Frag { v16h v; v4u q[2]; half_t h[16]; };

__device__ __forceinline__ float waveSum(float v) {
#pragma unroll
  for (int o = 16; o > 0; o >>= 1) v += __shfl_down(v, o, 32);
  return v;
}

// block-wide sum, result broadcast to all threads. blockDim.x multiple of 32.
__device__ float blockRed(float v, float* sm8) {
  int t = threadIdx.x;
  v = waveSum(v);
  if ((t & 31) == 0) sm8[t >> 5] = v;
  __syncthreads();
  float s = 0.f;
  if (t == 0) {
    int nw = blockDim.x >> 5;
    for (int i = 0; i < nw; ++i) s += sm8[i];
    sm8[0] = s;
  }
  __syncthreads();
  s = sm8[0];
  __syncthreads();
  return s;
}

__device__ __forceinline__ v8f wmma_f16(const Frag& a, const Frag& b, v8f c) {
  return __builtin_amdgcn_wmma_f32_16x16x32_f16(false, a.v, false, b.v,
                                                (short)0, c, false, false);
}

// ---------------------------------------------------------------------------
// TDM: async 2D tile load global -> LDS (Tensor Data Mover, D# per ISA ch.8).
// w = row length in elements (2-byte), h = #rows, strd = row stride (elems).
// Issued by one wave; caller must s_wait_tensorcnt + barrier before LDS reads.
// Toolchain builtin is the 6-arg variant:
//   (uint32x4 g0, int32x8 g1, int32x4 g2, int32x4 g3, int32x8 g4, i32 cpol)
// ---------------------------------------------------------------------------
#if __has_builtin(__builtin_amdgcn_tensor_load_to_lds)
#define HAVE_TDM 1
__device__ __forceinline__ void tdm_load_2d(unsigned lds_off, const void* g,
                                            unsigned w, unsigned h,
                                            unsigned strd) {
  unsigned long long ga = (unsigned long long)(size_t)g;
  // D# group0: count=1 | lds_addr | global_addr[56:0] | type=2 ("image")
  u32x4 g0 = { 1u, lds_off, (unsigned)ga,
               ((unsigned)(ga >> 32) & 0x01ffffffu) | (2u << 30) };
  // D# group1: data_size=1 (2B); tensor_dim0=w, tensor_dim1=h;
  //            tile_dim0=w, tile_dim1=h, tile_dim2=0; dim0_stride=strd.
  i32x8 g1 = { (int)0x00010000,
               (int)((w & 0xffffu) << 16),
               (int)(((w >> 16) & 0xffffu) | ((h & 0xffffu) << 16)),
               (int)(((h >> 16) & 0xffffu) | ((w & 0xffffu) << 16)),
               (int)(h & 0xffffu),
               (int)strd,
               0, 0 };
  i32x4 gz4 = { 0, 0, 0, 0 };
  i32x8 gz8 = { 0, 0, 0, 0, 0, 0, 0, 0 };
  __builtin_amdgcn_tensor_load_to_lds(g0, g1, gz4, gz4, gz8, 0);
}
#else
#define HAVE_TDM 0
#endif

__device__ __forceinline__ void tdm_wait0() {
#if __has_builtin(__builtin_amdgcn_s_wait_tensorcnt)
  __builtin_amdgcn_s_wait_tensorcnt(0);
#else
  asm volatile("s_wait_tensorcnt 0x0" ::: "memory");
#endif
}

// stage a [h x w] 2-byte-element panel into LDS (TDM if available)
__device__ __forceinline__ void stage_panel(half_t* lds, const half_t* g,
                                            unsigned w, unsigned h) {
#if HAVE_TDM
  if (threadIdx.x < 32) {
    tdm_load_2d((unsigned)(size_t)(void*)lds, (const void*)g, w, h, w);
    tdm_wait0();
  }
  __syncthreads();
#else
  for (unsigned i = threadIdx.x; i < w * h; i += blockDim.x) lds[i] = g[i];
  __syncthreads();
#endif
}

// ---------------------------------------------------------------------------
// K0: zero accumulators + output
// ---------------------------------------------------------------------------
__global__ void k_zero(float* dflat2, float* cs, float* vol, float* trsAs,
                       float* scal, float* dout) {
  int i = blockIdx.x * blockDim.x + threadIdx.x;
  if (i < Bb * Nn) dflat2[i] = 0.f;
  if (i < Bb * K1) cs[i] = 0.f;
  if (i < Bb) { vol[i] = 0.f; trsAs[i] = 0.f; }
  if (i < 64) scal[i] = 0.f;
  if (i < Bb * OUTC + 2) dout[i] = 0.f;
}

// ---------------------------------------------------------------------------
// K1: x1 = x@W_lin1+b ; s1 = tanh(x1@W_pool1+b) ; r1 = ||s1_n||^2 ; colsums
// ---------------------------------------------------------------------------
__global__ __launch_bounds__(32) void k_lin1(
    const float* __restrict__ x, const float* __restrict__ W1,
    const float* __restrict__ b1, const float* __restrict__ Wp,
    const float* __restrict__ bp, float* __restrict__ x1,
    half_t* __restrict__ x1h, float* __restrict__ s1,
    half_t* __restrict__ s1h, float* __restrict__ r1, float* __restrict__ cs) {
  int row = blockIdx.x;            // b*N + n
  int b = row >> 10;
  int lane = threadIdx.x;
  __shared__ float xr[Hh];
  const float* xp = x + (size_t)row * FIN;
  float acc = b1[lane];
  for (int k = 0; k < FIN; ++k) acc += xp[k] * W1[k * Hh + lane];
  x1[(size_t)row * Hh + lane] = acc;
  x1h[(size_t)row * Hh + lane] = (half_t)acc;
  xr[lane] = acc;
  __syncthreads();
  float rr = 0.f;
  for (int j = lane; j < K1; j += 32) {
    float z = bp[j];
    for (int k = 0; k < Hh; ++k) z += xr[k] * Wp[k * K1 + j];
    z = tanhf(z);
    half_t zh = (half_t)z;
    s1[(size_t)row * K1 + j] = z;
    s1h[(size_t)row * K1 + j] = zh;
    float zf = (float)zh;          // f16-rounded, consistent with Gram
    rr += zf * zf;
    atomicAdd(&cs[b * K1 + j], z);
  }
  rr = waveSum(rr);
  if (lane == 0) r1[row] = rr;
}

// ---------------------------------------------------------------------------
// K2: row sums of adj -> dflat, vol
// ---------------------------------------------------------------------------
__global__ __launch_bounds__(32) void k_rowsum(const float* __restrict__ adj,
                                               float* __restrict__ dflat,
                                               float* __restrict__ vol) {
  int row = blockIdx.x;
  int b = row >> 10;
  const float* ap = adj + (size_t)row * Nn;
  float s = 0.f;
  for (int j = threadIdx.x; j < Nn; j += 32) s += ap[j];
  s = waveSum(s);
  if (threadIdx.x == 0) { dflat[row] = s; atomicAdd(&vol[b], s); }
}

// ---------------------------------------------------------------------------
// K3: fused Gram + CT rewiring.  G = s1 s1^T (WMMA f16, K=64).
//     adj_new = sqrt(max(r_i+r_j-2G,0))/vol * adj   (stored f16)
//     trsAs[b] += sum(adj .* G) ; dflat2[b][i] += rowsum(adj_new)
// 64x64 tile per 128-thread block (4 waves, each 32x32 = 4 WMMA tiles)
// ---------------------------------------------------------------------------
__global__ __launch_bounds__(128) void k_gram(
    const half_t* __restrict__ s1h, const float* __restrict__ r1,
    const float* __restrict__ adj, const float* __restrict__ vol,
    half_t* __restrict__ adjn, float* __restrict__ dflat2,
    float* __restrict__ trsAs) {
  int b = blockIdx.z;
  int i0 = blockIdx.y * 64, j0 = blockIdx.x * 64;
  int wave = threadIdx.x >> 5, lane = threadIdx.x & 31;
  int g = lane >> 4, ln = lane & 15;
  int iw = i0 + (wave >> 1) * 32;
  int jw = j0 + (wave & 1) * 32;
  const half_t* S = s1h + (size_t)b * Nn * K1;

  v8f acc[2][2] = {{{}, {}}, {{}, {}}};
#pragma unroll
  for (int kc = 0; kc < K1; kc += 32) {
    Frag A[2], Bf[2];
#pragma unroll
    for (int t = 0; t < 2; ++t) {
      const half_t* pa = S + (size_t)(iw + t * 16 + ln) * K1 + kc + 8 * g;
      A[t].q[0] = *(const v4u*)pa;          // K = kc+8g .. +7
      A[t].q[1] = *(const v4u*)(pa + 16);   // K = kc+16+8g .. +7
      const half_t* pb = S + (size_t)(jw + t * 16 + ln) * K1 + kc + 8 * g;
      Bf[t].q[0] = *(const v4u*)pb;         // K = kc+8g .. +7
      Bf[t].q[1] = *(const v4u*)(pb + 16);  // K = kc+16+8g .. +7
    }
#pragma unroll
    for (int ti = 0; ti < 2; ++ti)
#pragma unroll
      for (int tj = 0; tj < 2; ++tj)
        acc[ti][tj] = wmma_f16(A[ti], Bf[tj], acc[ti][tj]);
  }

  float volb = vol[b] + (float)Nn * EPSf;   // trace(D) includes +EPS per diag
  float sAs = 0.f;
#pragma unroll
  for (int ti = 0; ti < 2; ++ti) {
#pragma unroll
    for (int tj = 0; tj < 2; ++tj) {
      int jcol = jw + tj * 16 + ln;
      float rj = r1[b * Nn + jcol];
#pragma unroll
      for (int r = 0; r < 8; ++r) {
        int i = iw + ti * 16 + r + 8 * g;   // C/D layout: m = r + 8*(lane/16)
        float gv = acc[ti][tj][r];
        float ri = r1[b * Nn + i];
        float d2 = ri + rj - 2.f * gv;
        float ct = sqrtf(fmaxf(d2, 0.f));
        float a = adj[((size_t)(b * Nn + i)) * Nn + jcol];
        float an = ct / volb * a;
        adjn[((size_t)(b * Nn + i)) * Nn + jcol] = (half_t)an;
        sAs += a * gv;
        float rs = an;                       // row-sum across 16-lane group
#pragma unroll
        for (int o = 1; o < 16; o <<= 1) rs += __shfl_xor(rs, o, 32);
        if (ln == 0) atomicAdd(&dflat2[b * Nn + i], rs);
      }
    }
  }
  float wv = waveSum(sAs);
  __shared__ float sm[4];
  if (lane == 0) sm[wave] = wv;
  __syncthreads();
  if (threadIdx.x == 0) atomicAdd(&trsAs[b], sm[0] + sm[1] + sm[2] + sm[3]);
}

// ---------------------------------------------------------------------------
// K4: ss1 = s1^T s1  [B,64,64]
// ---------------------------------------------------------------------------
__global__ __launch_bounds__(64) void k_ss1(const float* __restrict__ s1,
                                            float* __restrict__ ss1) {
  int k = blockIdx.x, b = blockIdx.y, l = threadIdx.x;
  const float* S = s1 + (size_t)b * Nn * K1;
  float v = 0.f;
  for (int n = 0; n < Nn; ++n) v += S[n * K1 + k] * S[n * K1 + l];
  ss1[((size_t)b * K1 + k) * K1 + l] = v;
}

// ---------------------------------------------------------------------------
// K5: ct_loss and ortho1 -> scal[0], scal[1]
// ---------------------------------------------------------------------------
__global__ __launch_bounds__(256) void k_ctred(
    const float* __restrict__ dflat, const float* __restrict__ r1,
    const float* __restrict__ cs, const float* __restrict__ trsAs,
    const float* __restrict__ ss1, float* __restrict__ scal) {
  __shared__ float sm8[8];
  int t = threadIdx.x;
  float ctacc = 0.f, osum = 0.f;
  for (int b = 0; b < Bb; ++b) {
    float p = 0.f;
    for (int n = t; n < Nn; n += 256)
      p += dflat[b * Nn + n] * r1[b * Nn + n];
    float trs = blockRed(p, sm8);
    float q = (t < K1) ? cs[b * K1 + t] * cs[b * K1 + t] : 0.f;
    float epst = EPSf * blockRed(q, sm8);   // EPS * ||colsum(s)||^2
    if (t == 0) {
      float num = trs + epst - trsAs[b];
      float den = trs + epst + EPSf;
      ctacc += num / den;
    }
    const float* SS = ss1 + (size_t)b * K1 * K1;
    float f = 0.f;
    for (int i = t; i < K1 * K1; i += 256) { float s = SS[i]; f += s * s; }
    float ssn = sqrtf(blockRed(f, sm8));
    float d = 0.f;
    for (int i = t; i < K1 * K1; i += 256) {
      int kk = i >> 6, ll = i & 63;
      float df = SS[i] / ssn - ((kk == ll) ? 1.f : 0.f);
      d += df * df;
    }
    float ds = blockRed(d, sm8);
    if (t == 0) osum += ds;
  }
  if (t == 0) { scal[0] = ctacc / (float)Bb; scal[1] = sqrtf(osum); }
}

// ---------------------------------------------------------------------------
// K6: y = adj_new @ x1   (WMMA f16, M=1024,N=32,K=1024).
// x1h panel (64KB) staged to LDS with TDM; B-frags read from LDS.
// ---------------------------------------------------------------------------
__global__ __launch_bounds__(128) void k_conv1(const half_t* __restrict__ adjn,
                                               const half_t* __restrict__ x1h,
                                               float* __restrict__ y) {
  __shared__ half_t Xs[Nn * Hh];           // 64 KB panel
  int b = blockIdx.y;
  int iw = blockIdx.x * 64 + (threadIdx.x >> 5) * 16;
  int lane = threadIdx.x & 31;
  int g = lane >> 4, ln = lane & 15;
  const half_t* Ab = adjn + (size_t)b * Nn * Nn;

  stage_panel(Xs, x1h + (size_t)b * Nn * Hh, Hh, Nn);

  v8f acc0 = {}, acc1 = {};
  for (int kc = 0; kc < Nn; kc += 32) {
    Frag A;
    const half_t* pa = Ab + (size_t)(iw + ln) * Nn + kc + 8 * g;
    A.q[0] = *(const v4u*)pa;
    A.q[1] = *(const v4u*)(pa + 16);
    __builtin_prefetch(pa + 32, 0, 1);
    Frag B0, B1;
#pragma unroll
    for (int e = 0; e < 16; ++e) {
      int k = kc + 8 * g + (e < 8 ? e : e + 8);
      B0.h[e] = Xs[k * Hh + ln];
      B1.h[e] = Xs[k * Hh + 16 + ln];
    }
    acc0 = wmma_f16(A, B0, acc0);
    acc1 = wmma_f16(A, B1, acc1);
  }
#pragma unroll
  for (int r = 0; r < 8; ++r) {
    int i = iw + r + 8 * g;
    y[((size_t)(b * Nn + i)) * Hh + ln] = acc0[r];
    y[((size_t)(b * Nn + i)) * Hh + 16 + ln] = acc1[r];
  }
}

// ---------------------------------------------------------------------------
// K7: x2 = y@Wrel1+brel1 + x1@Wroot1 ; s2 = softmax(x2@Wp2+bp2) ; r2
// ---------------------------------------------------------------------------
__global__ __launch_bounds__(32) void k_post1(
    const float* __restrict__ y, const float* __restrict__ x1,
    const float* __restrict__ Wrel1, const float* __restrict__ brel1,
    const float* __restrict__ Wroot1, const float* __restrict__ Wp2,
    const float* __restrict__ bp2, float* __restrict__ x2,
    float* __restrict__ s2, half_t* __restrict__ s2h,
    float* __restrict__ r2) {
  int row = blockIdx.x;
  int lane = threadIdx.x;
  __shared__ float yr[Hh], xr[Hh], x2r[Hh];
  yr[lane] = y[(size_t)row * Hh + lane];
  xr[lane] = x1[(size_t)row * Hh + lane];
  __syncthreads();
  float v = brel1[lane];
  for (int k = 0; k < Hh; ++k)
    v += yr[k] * Wrel1[k * Hh + lane] + xr[k] * Wroot1[k * Hh + lane];
  x2[(size_t)row * Hh + lane] = v;
  x2r[lane] = v;
  __syncthreads();
  float z = 0.f;
  if (lane < K2) {
    z = bp2[lane];
    for (int k = 0; k < Hh; ++k) z += x2r[k] * Wp2[k * K2 + lane];
  }
  float m = z;
#pragma unroll
  for (int o = 8; o > 0; o >>= 1) m = fmaxf(m, __shfl_xor(m, o, 16));
  float e = expf(z - m);
  float se = e;
#pragma unroll
  for (int o = 8; o > 0; o >>= 1) se += __shfl_xor(se, o, 16);
  float sv = e / se;
  float rr = 0.f;
  if (lane < K2) {
    s2[(size_t)row * K2 + lane] = sv;
    s2h[(size_t)row * K2 + lane] = (half_t)sv;
    rr = sv * sv;
  }
  rr = waveSum(rr);
  if (lane == 0) r2[row] = rr;
}

// ---------------------------------------------------------------------------
// K8: out = s2^T x2  [B,16,32]
// ---------------------------------------------------------------------------
__global__ __launch_bounds__(32) void k_out(const float* __restrict__ s2,
                                            const float* __restrict__ x2,
                                            float* __restrict__ outp) {
  int k = blockIdx.x, b = blockIdx.y, f = threadIdx.x;
  float s = 0.f;
  for (int n = 0; n < Nn; ++n)
    s += s2[((size_t)(b * Nn + n)) * K2 + k] *
         x2[((size_t)(b * Nn + n)) * Hh + f];
  outp[((size_t)b * K2 + k) * Hh + f] = s;
}

// ---------------------------------------------------------------------------
// K9: T = adj_new @ s2  (WMMA f16, N=16). s2h panel (32KB) staged via TDM.
// ---------------------------------------------------------------------------
__global__ __launch_bounds__(128) void k_T(const half_t* __restrict__ adjn,
                                           const half_t* __restrict__ s2h,
                                           float* __restrict__ T) {
  __shared__ half_t Ss[Nn * K2];           // 32 KB panel
  int b = blockIdx.y;
  int iw = blockIdx.x * 64 + (threadIdx.x >> 5) * 16;
  int lane = threadIdx.x & 31;
  int g = lane >> 4, ln = lane & 15;
  const half_t* Ab = adjn + (size_t)b * Nn * Nn;

  stage_panel(Ss, s2h + (size_t)b * Nn * K2, K2, Nn);

  v8f acc = {};
  for (int kc = 0; kc < Nn; kc += 32) {
    Frag A;
    const half_t* pa = Ab + (size_t)(iw + ln) * Nn + kc + 8 * g;
    A.q[0] = *(const v4u*)pa;
    A.q[1] = *(const v4u*)(pa + 16);
    Frag Bf;
#pragma unroll
    for (int e = 0; e < 16; ++e)
      Bf.h[e] = Ss[(kc + 8 * g + (e < 8 ? e : e + 8)) * K2 + ln];
    acc = wmma_f16(A, Bf, acc);
  }
#pragma unroll
  for (int r = 0; r < 8; ++r) {
    int i = iw + r + 8 * g;
    T[((size_t)(b * Nn + i)) * K2 + ln] = acc[r];
  }
}

// ---------------------------------------------------------------------------
// K10: poolA = s2^T T  [B,16,16]
// ---------------------------------------------------------------------------
__global__ __launch_bounds__(16) void k_pooladj(const float* __restrict__ s2,
                                                const float* __restrict__ T,
                                                float* __restrict__ poolA) {
  int k = blockIdx.x, b = blockIdx.y, l = threadIdx.x;
  float s = 0.f;
  for (int n = 0; n < Nn; ++n)
    s += s2[((size_t)(b * Nn + n)) * K2 + k] *
         T[((size_t)(b * Nn + n)) * K2 + l];
  poolA[((size_t)b * K2 + k) * K2 + l] = s;
}

// ---------------------------------------------------------------------------
// K11: per-batch tail: mincut loss pieces, ortho2, adj normalization,
//      pooled graph conv, readout MLP, log_softmax.
// ---------------------------------------------------------------------------
__global__ __launch_bounds__(256) void k_final(
    const float* __restrict__ poolA, const float* __restrict__ dflat2,
    const float* __restrict__ r2, const float* __restrict__ outp,
    const float* __restrict__ Wrel2, const float* __restrict__ brel2,
    const float* __restrict__ Wroot2, const float* __restrict__ Wlin2,
    const float* __restrict__ blin2, const float* __restrict__ Wlin3,
    const float* __restrict__ blin3, const float* __restrict__ s2,
    float* __restrict__ dout, float* __restrict__ scal) {
  int b = blockIdx.x, t = threadIdx.x;
  int k = t >> 4, l = t & 15;
  __shared__ float sm8[8];
  __shared__ float ss2[256], oa[256], oan[256], dk[16];
  __shared__ float tmp[512], x3[512], xs[32], x4[32], lg[16], sc[1];

  // ortho2 (k2=16): ss2 = s2^T s2
  const float* S2 = s2 + (size_t)b * Nn * K2;
  float v = 0.f;
  for (int n = 0; n < Nn; ++n) v += S2[n * K2 + k] * S2[n * K2 + l];
  ss2[t] = v;
  float fro = blockRed(v * v, sm8);
  float ssn = sqrtf(fro);
  float df = ss2[t] / ssn - ((k == l) ? 0.25f : 0.f);  // I/sqrt(16)
  float ortho2b = sqrtf(blockRed(df * df, sm8));

  // mincut num/den
  float num = blockRed((t < K2) ? poolA[((size_t)b * K2 + t) * K2 + t] : 0.f, sm8);
  float dp = 0.f;
  for (int n = t; n < Nn; n += 256)
    dp += (dflat2[b * Nn + n] + EPSf) * r2[b * Nn + n];
  float den = blockRed(dp, sm8);

  // zero diagonal + symmetric degree normalization of pooled adjacency
  float oav = poolA[((size_t)b * K2 + k) * K2 + l] * ((k == l) ? 0.f : 1.f);
  oa[t] = oav;
  __syncthreads();
  if (t < K2) {
    float rs = 0.f;
    for (int j = 0; j < K2; ++j) rs += oa[t * K2 + j];
    dk[t] = sqrtf(rs + EPSf) + EPSf;
  }
  __syncthreads();
  oan[t] = oa[t] / (dk[k] * dk[l]);
  __syncthreads();

  // pooled graph conv: x3 = (oan @ out)@Wrel2 + brel2 + out@Wroot2
  const float* Ob = outp + (size_t)b * K2 * Hh;
  for (int idx = t; idx < K2 * Hh; idx += 256) {
    int kk = idx >> 5, f = idx & 31;
    float s = 0.f;
    for (int j = 0; j < K2; ++j) s += oan[kk * K2 + j] * Ob[j * Hh + f];
    tmp[idx] = s;
  }
  __syncthreads();
  for (int idx = t; idx < K2 * Hh; idx += 256) {
    int kk = idx >> 5, f = idx & 31;
    float s = brel2[f];
    for (int gg = 0; gg < Hh; ++gg)
      s += tmp[kk * Hh + gg] * Wrel2[gg * Hh + f] +
           Ob[kk * Hh + gg] * Wroot2[gg * Hh + f];
    x3[idx] = s;
  }
  __syncthreads();
  // readout: sum over nodes, relu(lin2), lin3, log_softmax
  if (t < Hh) {
    float s = 0.f;
    for (int kk = 0; kk < K2; ++kk) s += x3[kk * Hh + t];
    xs[t] = s;
  }
  __syncthreads();
  if (t < Hh) {
    float s = blin2[t];
    for (int gg = 0; gg < Hh; ++gg) s += xs[gg] * Wlin2[gg * Hh + t];
    x4[t] = fmaxf(s, 0.f);
  }
  __syncthreads();
  if (t < OUTC) {
    float s = blin3[t];
    for (int gg = 0; gg < Hh; ++gg) s += x4[gg] * Wlin3[gg * OUTC + t];
    lg[t] = s;
  }
  __syncthreads();
  if (t == 0) {
    float mx = lg[0];
    for (int o = 1; o < OUTC; ++o) mx = fmaxf(mx, lg[o]);
    float se = 0.f;
    for (int o = 0; o < OUTC; ++o) se += expf(lg[o] - mx);
    sc[0] = logf(se) + mx;
    scal[2 + b] = (-(num / den) + ortho2b) * (1.0f / (float)Bb);
  }
  __syncthreads();
  if (t < OUTC) dout[b * OUTC + t] = lg[t] - sc[0];
}

// ---------------------------------------------------------------------------
// K12: assemble scalar outputs deterministically
// ---------------------------------------------------------------------------
__global__ void k_scalars(const float* __restrict__ scal,
                          float* __restrict__ dout) {
  if (threadIdx.x == 0 && blockIdx.x == 0) {
    dout[Bb * OUTC] = scal[0] + scal[1];   // ct_loss + ortho1
    float s = 0.f;
    for (int b = 0; b < Bb; ++b) s += scal[2 + b];
    dout[Bb * OUTC + 1] = s;               // mincut + ortho2
  }
}

// ---------------------------------------------------------------------------
extern "C" void kernel_launch(void* const* d_in, const int* in_sizes, int n_in,
                              void* d_out, int out_size, void* d_ws,
                              size_t ws_size, hipStream_t stream) {
  const float* x     = (const float*)d_in[0];
  const float* adj   = (const float*)d_in[1];
  // d_in[2] = mask (all ones) -- intentionally unused
  const float* Wlin1 = (const float*)d_in[3];
  const float* blin1 = (const float*)d_in[4];
  const float* Wp1   = (const float*)d_in[5];
  const float* bp1   = (const float*)d_in[6];
  const float* Wp2   = (const float*)d_in[7];
  const float* bp2   = (const float*)d_in[8];
  const float* Wrel1 = (const float*)d_in[9];
  const float* brel1 = (const float*)d_in[10];
  const float* Wroot1= (const float*)d_in[11];
  const float* Wrel2 = (const float*)d_in[12];
  const float* brel2 = (const float*)d_in[13];
  const float* Wroot2= (const float*)d_in[14];
  const float* Wlin2 = (const float*)d_in[15];
  const float* blin2 = (const float*)d_in[16];
  const float* Wlin3 = (const float*)d_in[17];
  const float* blin3 = (const float*)d_in[18];
  float* dout = (float*)d_out;

  char* w = (char*)d_ws;
  auto alloc = [&](size_t bytes) -> void* {
    void* p = (void*)w;
    w += (bytes + 255) & ~(size_t)255;
    return p;
  };
  float*  x1     = (float*) alloc((size_t)Bb * Nn * Hh * 4);
  half_t* x1h    = (half_t*)alloc((size_t)Bb * Nn * Hh * 2);
  float*  s1     = (float*) alloc((size_t)Bb * Nn * K1 * 4);
  half_t* s1h    = (half_t*)alloc((size_t)Bb * Nn * K1 * 2);
  float*  r1     = (float*) alloc((size_t)Bb * Nn * 4);
  float*  cs     = (float*) alloc((size_t)Bb * K1 * 4);
  float*  dflat  = (float*) alloc((size_t)Bb * Nn * 4);
  float*  vol    = (float*) alloc((size_t)Bb * 4);
  float*  trsAs  = (float*) alloc((size_t)Bb * 4);
  float*  dflat2 = (float*) alloc((size_t)Bb * Nn * 4);
  half_t* adjn   = (half_t*)alloc((size_t)Bb * Nn * Nn * 2);
  float*  y      = (float*) alloc((size_t)Bb * Nn * Hh * 4);
  float*  x2     = (float*) alloc((size_t)Bb * Nn * Hh * 4);
  float*  s2     = (float*) alloc((size_t)Bb * Nn * K2 * 4);
  half_t* s2h    = (half_t*)alloc((size_t)Bb * Nn * K2 * 2);
  float*  r2     = (float*) alloc((size_t)Bb * Nn * 4);
  float*  outp   = (float*) alloc((size_t)Bb * K2 * Hh * 4);
  float*  T      = (float*) alloc((size_t)Bb * Nn * K2 * 4);
  float*  poolA  = (float*) alloc((size_t)Bb * K2 * K2 * 4);
  float*  ss1    = (float*) alloc((size_t)Bb * K1 * K1 * 4);
  float*  scal   = (float*) alloc(64 * 4);

  k_zero<<<64, 256, 0, stream>>>(dflat2, cs, vol, trsAs, scal, dout);
  k_lin1<<<Bb * Nn, 32, 0, stream>>>(x, Wlin1, blin1, Wp1, bp1, x1, x1h, s1,
                                     s1h, r1, cs);
  k_rowsum<<<Bb * Nn, 32, 0, stream>>>(adj, dflat, vol);
  k_gram<<<dim3(Nn / 64, Nn / 64, Bb), 128, 0, stream>>>(s1h, r1, adj, vol,
                                                         adjn, dflat2, trsAs);
  k_ss1<<<dim3(K1, Bb), 64, 0, stream>>>(s1, ss1);
  k_ctred<<<1, 256, 0, stream>>>(dflat, r1, cs, trsAs, ss1, scal);
  k_conv1<<<dim3(Nn / 64, Bb), 128, 0, stream>>>(adjn, x1h, y);
  k_post1<<<Bb * Nn, 32, 0, stream>>>(y, x1, Wrel1, brel1, Wroot1, Wp2, bp2,
                                      x2, s2, s2h, r2);
  k_out<<<dim3(K2, Bb), 32, 0, stream>>>(s2, x2, outp);
  k_T<<<dim3(Nn / 64, Bb), 128, 0, stream>>>(adjn, s2h, T);
  k_pooladj<<<dim3(K2, Bb), 16, 0, stream>>>(s2, T, poolA);
  k_final<<<Bb, 256, 0, stream>>>(poolA, dflat2, r2, outp, Wrel2, brel2,
                                  Wroot2, Wlin2, blin2, Wlin3, blin3, s2,
                                  dout, scal);
  k_scalars<<<1, 32, 0, stream>>>(scal, dout);
  (void)in_sizes; (void)n_in; (void)out_size; (void)ws_size;
}
